// NaryTreeLSTMCell_67138928771802
// MI455X (gfx1250) — matrix operand order repaired and checked by reading, not compiled, not run on hardware
//
#include <hip/hip_runtime.h>
#include <hip/hip_bf16.h>
#include <math.h>

typedef __attribute__((ext_vector_type(16))) __bf16 v16bf;
typedef __attribute__((ext_vector_type(8)))  float  v8f;

// Problem dims
constexpr int Bn = 32, Ln = 512, En = 1024, Hn = 1024;
constexpr int M  = Bn * Ln;      // 16384 rows (B*L)
constexpr int K  = 1024;         // E == H
constexpr int KB = K / 32;       // 32 K-fragments of depth 32
constexpr int N1 = 4096;         // GEMM1 cols: [iou_x 3072 | fx 1024]
constexpr int N2 = 8192;         // GEMM2 cols: [iou_r 3072 | iou_l 3072 | fh_r 1024 | fh_l 1024]

// Workspace layout (bytes)
constexpr size_t OFF_XA    = 0;                                   // x   bf16 frag-major  32MB
constexpr size_t OFF_HA    = OFF_XA  + (size_t)M * K * 2;         // h0  bf16 frag-major  32MB
constexpr size_t OFF_B1    = OFF_HA  + (size_t)M * K * 2;         // B1  bf16 frag-major   8MB
constexpr size_t OFF_B2    = OFF_B1  + (size_t)K * N1 * 2;        // B2  bf16 frag-major  16MB
constexpr size_t OFF_IOU   = OFF_B2  + (size_t)K * N2 * 2;        // iou f32 [M,3072]    192MB
constexpr size_t OFF_FX    = OFF_IOU + (size_t)M * 3072 * 4;      // fx  f32 [M,1024]     64MB
constexpr size_t OFF_FHR   = OFF_FX  + (size_t)M * 1024 * 4;      // fh_r f32             64MB
constexpr size_t OFF_FHL   = OFF_FHR + (size_t)M * 1024 * 4;      // fh_l f32             64MB
constexpr size_t OFF_FLAGS = OFF_FHL + (size_t)M * 1024 * 4;      // updated flags [M]

__device__ __forceinline__ float sigmoidf_(float x) {
    return 1.0f / (1.0f + __expf(-x));
}

// ---------------------------------------------------------------------------
// A-operand convert: f32 row-major [M][K] -> bf16 WMMA A-fragment-major.
// Fragment = (mb, kb); 32 lanes x 16 bf16 (32B) contiguous per lane.
// Lane<16: rows m=mb*16+lane, K=j (0..15)
// Lane>=16: rows m=mb*16+lane-16, K=16+j (16..31)
// ---------------------------------------------------------------------------
__global__ void k_conv_a(const float* __restrict__ src, __bf16* __restrict__ dst) {
    int idx = blockIdx.x * blockDim.x + threadIdx.x;
    if (idx >= M * K) return;
    int j    = idx & 15;
    int lane = (idx >> 4) & 31;
    int t    = idx >> 9;
    int kb   = t % KB;
    int mb   = t / KB;
    int m = mb * 16 + (lane & 15);
    int k;
    if (lane < 16) k = kb * 32 + j;
    else           k = kb * 32 + 16 + j;
    dst[idx] = (__bf16)src[(size_t)m * K + k];
}

// ---------------------------------------------------------------------------
// B-operand convert for GEMM1: cols [0,3072)=W_ioux, [3072,4096)=W_fx.
// B fragment (nb, kb): lane<16 -> n=nb*16+lane, K=kb*32+j (j=0..15);
//                      lane>=16 -> n=nb*16+lane-16, K=kb*32+16+j.
// ---------------------------------------------------------------------------
__global__ void k_conv_b1(const float* __restrict__ W_ioux,
                          const float* __restrict__ W_fx,
                          __bf16* __restrict__ dst) {
    int idx = blockIdx.x * blockDim.x + threadIdx.x;
    if (idx >= K * N1) return;
    int j    = idx & 15;
    int lane = (idx >> 4) & 31;
    int t    = idx >> 9;
    int kb   = t % KB;
    int nb   = t / KB;
    int n = nb * 16 + (lane & 15);
    int k = kb * 32 + ((lane < 16) ? j : (16 + j));
    float v;
    if (n < 3072) v = W_ioux[(size_t)k * 3072 + n];
    else          v = W_fx[(size_t)k * 1024 + (n - 3072)];
    dst[idx] = (__bf16)v;
}

// GEMM2 B: [iouh_r | iouh_l | (fh0+fh1) | (fh2+fh3)]
__global__ void k_conv_b2(const float* __restrict__ W_iouh_r,
                          const float* __restrict__ W_iouh_l,
                          const float* __restrict__ W_fh0,
                          const float* __restrict__ W_fh1,
                          const float* __restrict__ W_fh2,
                          const float* __restrict__ W_fh3,
                          __bf16* __restrict__ dst) {
    int idx = blockIdx.x * blockDim.x + threadIdx.x;
    if (idx >= K * N2) return;
    int j    = idx & 15;
    int lane = (idx >> 4) & 31;
    int t    = idx >> 9;
    int kb   = t % KB;
    int nb   = t / KB;
    int n = nb * 16 + (lane & 15);
    int k = kb * 32 + ((lane < 16) ? j : (16 + j));
    float v;
    if      (n < 3072) v = W_iouh_r[(size_t)k * 3072 + n];
    else if (n < 6144) v = W_iouh_l[(size_t)k * 3072 + (n - 3072)];
    else if (n < 7168) v = W_fh0[(size_t)k * 1024 + (n - 6144)] +
                           W_fh1[(size_t)k * 1024 + (n - 6144)];
    else               v = W_fh2[(size_t)k * 1024 + (n - 7168)] +
                           W_fh3[(size_t)k * 1024 + (n - 7168)];
    dst[idx] = (__bf16)v;
}

// ---------------------------------------------------------------------------
// GEMM1: O = x_bf16 @ [W_ioux|W_fx] ; epilogue adds biases, splits dest.
// Block = 8 waves (2M x 4N), wave tile 64x32 (4x2 WMMA), block tile 128x128.
// Operands stream fragment-major from L2 (all operands fit in 192MB L2).
// ---------------------------------------------------------------------------
__global__ __launch_bounds__(256) void k_gemm1(
    const __bf16* __restrict__ A, const __bf16* __restrict__ Bm,
    float* __restrict__ iou, float* __restrict__ fx,
    const float* __restrict__ b_ioux, const float* __restrict__ b_fx) {
    const int lane = threadIdx.x & 31;
    const int w    = threadIdx.x >> 5;
    const int wm = w & 1, wn = w >> 1;
    const int m0 = blockIdx.x * 128 + wm * 64;
    const int n0 = blockIdx.y * 128 + wn * 32;
    const int mb0 = m0 >> 4, nb0 = n0 >> 4;
    const v16bf* __restrict__ Af = (const v16bf*)A;
    const v16bf* __restrict__ Bf = (const v16bf*)Bm;

    v8f acc[4][2];
#pragma unroll
    for (int i = 0; i < 4; ++i)
#pragma unroll
        for (int jj = 0; jj < 2; ++jj) acc[i][jj] = (v8f)0.0f;

    for (int kb = 0; kb < KB; ++kb) {
        v16bf a[4], b[2];
#pragma unroll
        for (int i = 0; i < 4; ++i)
            a[i] = Af[((size_t)(mb0 + i) * KB + kb) * 32 + lane];
#pragma unroll
        for (int jj = 0; jj < 2; ++jj)
            b[jj] = Bf[((size_t)(nb0 + jj) * KB + kb) * 32 + lane];
        if (kb + 1 < KB) {   // -> global_prefetch_b8 (gfx1250)
            __builtin_prefetch(&Af[((size_t)mb0 * KB + kb + 1) * 32 + lane], 0, 3);
            __builtin_prefetch(&Bf[((size_t)nb0 * KB + kb + 1) * 32 + lane], 0, 3);
        }
#pragma unroll
        for (int i = 0; i < 4; ++i)
#pragma unroll
            for (int jj = 0; jj < 2; ++jj)
                acc[i][jj] = __builtin_amdgcn_wmma_f32_16x16x32_bf16(
                    false, a[i], false, b[jj], (short)0, acc[i][jj], false, false);
    }

    const int lm = lane & 15;
    const int rb = (lane >> 4) * 8;   // hi half-lanes hold rows 8..15
#pragma unroll
    for (int i = 0; i < 4; ++i)
#pragma unroll
        for (int jj = 0; jj < 2; ++jj)
#pragma unroll
            for (int r = 0; r < 8; ++r) {
                int m = m0 + i * 16 + rb + r;
                int n = n0 + jj * 16 + lm;
                float v = acc[i][jj][r];
                if (n < 3072) iou[(size_t)m * 3072 + n] = v + b_ioux[n];
                else          fx[(size_t)m * 1024 + (n - 3072)] = v + b_fx[n - 3072];
            }
}

// ---------------------------------------------------------------------------
// GEMM2: O = h0_bf16 @ [W_iouh_r|W_iouh_l|W_fh_r|W_fh_l].
// iou_r/iou_l parts scatter-add (atomic) into iou at tree_ids_dr/dl rows;
// fh parts stored dense for later gather.
// ---------------------------------------------------------------------------
__global__ __launch_bounds__(256) void k_gemm2(
    const __bf16* __restrict__ A, const __bf16* __restrict__ Bm,
    float* __restrict__ iou, float* __restrict__ fhr, float* __restrict__ fhl,
    const int* __restrict__ idr, const int* __restrict__ idl) {
    const int lane = threadIdx.x & 31;
    const int w    = threadIdx.x >> 5;
    const int wm = w & 1, wn = w >> 1;
    const int m0 = blockIdx.x * 128 + wm * 64;
    const int n0 = blockIdx.y * 128 + wn * 32;
    const int mb0 = m0 >> 4, nb0 = n0 >> 4;
    const v16bf* __restrict__ Af = (const v16bf*)A;
    const v16bf* __restrict__ Bf = (const v16bf*)Bm;

    v8f acc[4][2];
#pragma unroll
    for (int i = 0; i < 4; ++i)
#pragma unroll
        for (int jj = 0; jj < 2; ++jj) acc[i][jj] = (v8f)0.0f;

    for (int kb = 0; kb < KB; ++kb) {
        v16bf a[4], b[2];
#pragma unroll
        for (int i = 0; i < 4; ++i)
            a[i] = Af[((size_t)(mb0 + i) * KB + kb) * 32 + lane];
#pragma unroll
        for (int jj = 0; jj < 2; ++jj)
            b[jj] = Bf[((size_t)(nb0 + jj) * KB + kb) * 32 + lane];
        if (kb + 1 < KB) {
            __builtin_prefetch(&Af[((size_t)mb0 * KB + kb + 1) * 32 + lane], 0, 3);
            __builtin_prefetch(&Bf[((size_t)nb0 * KB + kb + 1) * 32 + lane], 0, 3);
        }
#pragma unroll
        for (int i = 0; i < 4; ++i)
#pragma unroll
            for (int jj = 0; jj < 2; ++jj)
                acc[i][jj] = __builtin_amdgcn_wmma_f32_16x16x32_bf16(
                    false, a[i], false, b[jj], (short)0, acc[i][jj], false, false);
    }

    const int lm = lane & 15;
    const int rb = (lane >> 4) * 8;
#pragma unroll
    for (int i = 0; i < 4; ++i)
#pragma unroll
        for (int jj = 0; jj < 2; ++jj)
#pragma unroll
            for (int r = 0; r < 8; ++r) {
                int m = m0 + i * 16 + rb + r;
                int n = n0 + jj * 16 + lm;
                float v = acc[i][jj][r];
                if (n < 6144) {
                    const int* ids = (n < 3072) ? idr : idl;
                    int nn = (n < 3072) ? n : (n - 3072);
                    int bb = m >> 9;                 // batch = m / L
                    int tgt = ids[m];                // node within batch
                    atomicAdd(&iou[((size_t)(bb * Ln + tgt)) * 3072 + nn], v);
                } else if (n < 7168) {
                    fhr[(size_t)m * 1024 + (n - 6144)] = v;
                } else {
                    fhl[(size_t)m * 1024 + (n - 7168)] = v;
                }
            }
}

// ----------------------- pointwise / scatter tail --------------------------
__global__ void k_zero_flags(int* __restrict__ flags) {
    int i = blockIdx.x * blockDim.x + threadIdx.x;
    if (i < M) flags[i] = 0;
}

__global__ void k_scatter_flags(const int* __restrict__ id_d, int* __restrict__ flags) {
    int i = blockIdx.x * blockDim.x + threadIdx.x;
    if (i >= M) return;
    int id = id_d[i];
    if (id != 0) flags[(i >> 9 << 9) + id] = 1;   // flags[b*L + id]; node 0 never updated
}

// c_pre = sigmoid(i) * tanh(u), written into output c region (accumulator)
__global__ void k_cpre(const float* __restrict__ iou, float* __restrict__ c_out) {
    int e = blockIdx.x * blockDim.x + threadIdx.x;
    if (e >= M * Hn) return;
    int m = e >> 10, h = e & 1023;
    float iv = sigmoidf_(iou[(size_t)m * 3072 + h]);
    float uv = tanhf(iou[(size_t)m * 3072 + 2048 + h]);
    c_out[e] = iv * uv;
}

// f = sigmoid(fx[id_d] + fh_r[idr] + fh_l[idl]); c_out[id_d] += f * c0
__global__ void k_scatter_fc(const float* __restrict__ fx,
                             const float* __restrict__ fhr,
                             const float* __restrict__ fhl,
                             const float* __restrict__ c0,
                             const int* __restrict__ id_d,
                             const int* __restrict__ idr,
                             const int* __restrict__ idl,
                             float* __restrict__ c_out) {
    int e = blockIdx.x * blockDim.x + threadIdx.x;
    if (e >= M * Hn) return;
    int m = e >> 10, h = e & 1023;
    int b0 = (m >> 9) << 9;                       // b * L
    int nd = id_d[m], nr = idr[m], nl = idl[m];
    float f = fx[((size_t)(b0 + nd)) * 1024 + h] +
              fhr[((size_t)(b0 + nr)) * 1024 + h] +
              fhl[((size_t)(b0 + nl)) * 1024 + h];
    f = sigmoidf_(f);
    atomicAdd(&c_out[((size_t)(b0 + nd)) * 1024 + h], f * c0[e]);
}

// h = sigmoid(o) * tanh(c); masked select vs h0/c0
__global__ void k_final(const float* __restrict__ iou,
                        const float* __restrict__ h0,
                        const float* __restrict__ c0,
                        const int* __restrict__ flags,
                        float* __restrict__ h_out,
                        float* __restrict__ c_out) {
    int e = blockIdx.x * blockDim.x + threadIdx.x;
    if (e >= M * Hn) return;
    int m = e >> 10, h = e & 1023;
    float c  = c_out[e];
    float o  = sigmoidf_(iou[(size_t)m * 3072 + 1024 + h]);
    float hv = o * tanhf(c);
    bool upd = flags[m] != 0;
    h_out[e] = upd ? hv : h0[e];
    c_out[e] = upd ? c  : c0[e];
}

// ---------------------------------------------------------------------------
extern "C" void kernel_launch(void* const* d_in, const int* in_sizes, int n_in,
                              void* d_out, int out_size, void* d_ws, size_t ws_size,
                              hipStream_t stream) {
    const float* x        = (const float*)d_in[0];
    const float* h0       = (const float*)d_in[1];
    const float* c0       = (const float*)d_in[2];
    const float* W_ioux   = (const float*)d_in[3];
    const float* b_ioux   = (const float*)d_in[4];
    const float* W_iouh_r = (const float*)d_in[5];
    const float* W_iouh_l = (const float*)d_in[6];
    const float* W_fx     = (const float*)d_in[7];
    const float* b_fx     = (const float*)d_in[8];
    const float* W_fh0    = (const float*)d_in[9];
    const float* W_fh1    = (const float*)d_in[10];
    const float* W_fh2    = (const float*)d_in[11];
    const float* W_fh3    = (const float*)d_in[12];
    const int*   id_d     = (const int*)d_in[13];
    const int*   id_dr    = (const int*)d_in[14];
    const int*   id_dl    = (const int*)d_in[15];

    char* ws = (char*)d_ws;
    __bf16* xA    = (__bf16*)(ws + OFF_XA);
    __bf16* hA    = (__bf16*)(ws + OFF_HA);
    __bf16* B1w   = (__bf16*)(ws + OFF_B1);
    __bf16* B2w   = (__bf16*)(ws + OFF_B2);
    float*  iou   = (float*)(ws + OFF_IOU);
    float*  fx    = (float*)(ws + OFF_FX);
    float*  fhr   = (float*)(ws + OFF_FHR);
    float*  fhl   = (float*)(ws + OFF_FHL);
    int*    flags = (int*)(ws + OFF_FLAGS);

    float* h_out = (float*)d_out;
    float* c_out = (float*)d_out + (size_t)M * Hn;

    const int T = 256;

    // Operand conversion / pre-swizzle into WMMA fragment layout
    k_conv_a <<<(M * K) / T, T, 0, stream>>>(x,  xA);
    k_conv_a <<<(M * K) / T, T, 0, stream>>>(h0, hA);
    k_conv_b1<<<(K * N1) / T, T, 0, stream>>>(W_ioux, W_fx, B1w);
    k_conv_b2<<<(K * N2) / T, T, 0, stream>>>(W_iouh_r, W_iouh_l,
                                              W_fh0, W_fh1, W_fh2, W_fh3, B2w);

    // updated-node flags
    k_zero_flags   <<<M / T, T, 0, stream>>>(flags);
    k_scatter_flags<<<M / T, T, 0, stream>>>(id_d, flags);

    // GEMM1: iou_x (+bias) and fx_out (+bias)
    k_gemm1<<<dim3(M / 128, N1 / 128), T, 0, stream>>>(xA, B1w, iou, fx, b_ioux, b_fx);

    // GEMM2: child-h contributions scatter into iou; fh_r / fh_l stored dense
    k_gemm2<<<dim3(M / 128, N2 / 128), T, 0, stream>>>(hA, B2w, iou, fhr, fhl,
                                                       id_dr, id_dl);

    // Gate nonlinearities + tree scatter of f*c0 + final masked outputs
    k_cpre      <<<(M * Hn) / T, T, 0, stream>>>(iou, c_out);
    k_scatter_fc<<<(M * Hn) / T, T, 0, stream>>>(fx, fhr, fhl, c0,
                                                 id_d, id_dr, id_dl, c_out);
    k_final     <<<(M * Hn) / T, T, 0, stream>>>(iou, h0, c0, flags, h_out, c_out);
}
